// NodeClassificationGCNN_16509854286320
// MI455X (gfx1250) — matrix operand order repaired and checked by reading, not compiled, not run on hardware
//
#include <hip/hip_runtime.h>
#include <hip/hip_bf16.h>
#include <math.h>

// ---------------------------------------------------------------------------
// GCN forward on MI455X (gfx1250, wave32).
//   h0  = X @ W1                   (bf16 WMMA, f32 accum, async-LDS staging)
//   a1  = segment_sum(edges, h0)   (edge-parallel atomicAdd, L2-resident)
//   h1  = relu(a1)                 (fused into GEMM2 fragment build)
//   h2  = h1 @ W2                  (bf16 WMMA, f32 accum, N=40 padded to 48)
//   a2  = segment_sum(edges, h2)
//   out = log_softmax(relu(a2))
// Workspace: bufA = N*256 f32, bufB = N*256 f32  (~205 MB total)
// ---------------------------------------------------------------------------

#define N_NODES 100000
#define F_IN    512
#define HIDDEN  256
#define NCLASS  40

typedef __attribute__((ext_vector_type(16))) __bf16 v16bf;
typedef __attribute__((ext_vector_type(4)))  __bf16 v4bf;
typedef __attribute__((ext_vector_type(8)))  float  v8f;
typedef __attribute__((ext_vector_type(4)))  float  v4f;

// Async copy of one 16B chunk global->LDS (GVS mode), tracked by ASYNCcnt.
__device__ __forceinline__ void async_b128_to_lds(unsigned lds_off, unsigned gvoff,
                                                  const float* sbase) {
    asm volatile("global_load_async_to_lds_b128 %0, %1, %2 offset:0"
                 :: "v"(lds_off), "v"(gvoff), "s"(sbase) : "memory");
}
__device__ __forceinline__ void wait_asynccnt0() {
    asm volatile("s_wait_asynccnt 0x0" ::: "memory");
}

// Build a v16bf A-fragment from f32 LDS in contiguous k order (same as B):
// lane l: M=l%16, halves 0..15 = K kA0..kA0+15  (kA0=(l/16)*16)
template<bool RELU>
__device__ __forceinline__ v16bf build_a_frag(const float* ap, int kA0) {
    v4f x0 = *(const v4f*)(ap + kA0);
    v4f x1 = *(const v4f*)(ap + kA0 + 4);
    v4f x2 = *(const v4f*)(ap + kA0 + 8);
    v4f x3 = *(const v4f*)(ap + kA0 + 12);
    v16bf a;
    #pragma unroll
    for (int j = 0; j < 4; ++j) {
        float f0 = x0[j], f1 = x1[j], f2 = x2[j], f3 = x3[j];
        if (RELU) { f0 = fmaxf(f0, 0.f); f1 = fmaxf(f1, 0.f);
                    f2 = fmaxf(f2, 0.f); f3 = fmaxf(f3, 0.f); }
        a[j]      = (__bf16)f0;
        a[4 + j]  = (__bf16)f1;
        a[8 + j]  = (__bf16)f2;
        a[12 + j] = (__bf16)f3;
    }
    return a;
}

// ---------------------------------------------------------------------------
// GEMM1: C[N_NODES,256] = X[N_NODES,512] @ W1[512,256]
// Block: 256 thr (8 waves), tile BM=64 x BN=256, K step 32.
// Wave w: m-subtile w%4, n-half w/4 (8 n-subtiles of 16 -> 8 accumulators).
// GUARD=false: full blocks, async-LDS A staging. GUARD=true: tail block.
// ---------------------------------------------------------------------------
template<bool GUARD>
__global__ __launch_bounds__(256)
void gemm1_xw1_wmma(const float* __restrict__ X, const float* __restrict__ W1,
                    float* __restrict__ C, int block0) {
    __shared__ __attribute__((aligned(16))) float  AsubF[64 * 32];   // f32 [m][k]
    __shared__ __attribute__((aligned(32))) __bf16 Bt[256 * 32];     // bf16 [n][k]

    const int t    = threadIdx.x;
    const int wave = t >> 5;
    const int lane = t & 31;
    const int m0   = (block0 + (int)blockIdx.x) * 64;

    const int msub  = wave & 3;
    const int nhalf = wave >> 2;
    const int lmod  = lane & 15;
    const int lhi   = lane >> 4;
    const int kA0   = lhi * 16;
    const int kB0   = lhi * 16;

    v8f acc[8] = {};

    for (int kt = 0; kt < F_IN; kt += 32) {
        // ---- stage A tile 64x32 f32: 512 16B chunks, 2 per thread ----
        #pragma unroll
        for (int i = 0; i < 2; ++i) {
            int slot = t + i * 256;         // 0..511
            int r = slot >> 3, cg = slot & 7;
            if (!GUARD) {
                unsigned lds = (unsigned)(uintptr_t)&AsubF[r * 32 + cg * 4];
                unsigned go  = (unsigned)(((unsigned)(m0 + r) * F_IN + kt + cg * 4)
                                          * sizeof(float));
                async_b128_to_lds(lds, go, X);
            } else {
                int gm = m0 + r;
                v4f v = {};
                if (gm < N_NODES) v = *(const v4f*)&X[(size_t)gm * F_IN + kt + cg * 4];
                *(v4f*)&AsubF[r * 32 + cg * 4] = v;
            }
        }
        // ---- stage B tile 32x256 transposed -> Bt[n][k] (overlaps async A) ----
        #pragma unroll
        for (int i = 0; i < 8; ++i) {
            int slot = t + i * 256;         // 0..2047 float4 slots
            int k = slot >> 6, ng = slot & 63;
            v4f v = *(const v4f*)&W1[(size_t)(kt + k) * HIDDEN + ng * 4];
            #pragma unroll
            for (int j = 0; j < 4; ++j)
                Bt[(ng * 4 + j) * 32 + k] = (__bf16)v[j];
        }
        if (!GUARD) wait_asynccnt0();
        __syncthreads();

        v16bf a = build_a_frag<false>(&AsubF[(msub * 16 + lmod) * 32], kA0);
        #pragma unroll
        for (int j = 0; j < 8; ++j) {
            int n = nhalf * 128 + j * 16 + lmod;
            v16bf b = *(const v16bf*)&Bt[n * 32 + kB0];   // 32B contiguous
            acc[j] = __builtin_amdgcn_wmma_f32_16x16x32_bf16(
                false, a, false, b, (short)0, acc[j], false, false);
        }
        __syncthreads();
    }

    // store: C/D layout: VGPR r, lane l: row = (l/16)*8 + r, col = l%16
    const int rbase = m0 + msub * 16 + lhi * 8;
    #pragma unroll
    for (int j = 0; j < 8; ++j) {
        int n = nhalf * 128 + j * 16 + lmod;
        #pragma unroll
        for (int r = 0; r < 8; ++r) {
            int gm = rbase + r;
            if (!GUARD || gm < N_NODES) C[(size_t)gm * HIDDEN + n] = acc[j][r];
        }
    }
}

// ---------------------------------------------------------------------------
// GEMM2: C[N_NODES,40] = relu(H[N_NODES,256]) @ W2[256,40]   (N padded to 48)
// Block: 256 thr, BM=128; wave w owns m-subtile w; 3 n-subtiles of 16.
// ReLU fused into fragment build. W2 preloaded to LDS once.
// ---------------------------------------------------------------------------
template<bool GUARD>
__global__ __launch_bounds__(256)
void gemm2_hw2_wmma(const float* __restrict__ H, const float* __restrict__ W2,
                    float* __restrict__ C, int block0) {
    __shared__ __attribute__((aligned(16))) float  AsubF[128 * 32];  // f32 [m][k]
    __shared__ __attribute__((aligned(32))) __bf16 Bt[48 * 256];     // bf16 [n][k]

    const int t    = threadIdx.x;
    const int wave = t >> 5;
    const int lane = t & 31;
    const int m0   = (block0 + (int)blockIdx.x) * 128;
    const int lmod = lane & 15;
    const int lhi  = lane >> 4;
    const int kA0  = lhi * 16;
    const int kB0  = lhi * 16;

    // preload Bt[48][256]: thread t supplies k=t (reads its contiguous W2 row)
    {
        const float* w2r = &W2[(size_t)t * NCLASS];
        #pragma unroll
        for (int n = 0; n < NCLASS; ++n) Bt[n * 256 + t] = (__bf16)w2r[n];
        #pragma unroll
        for (int n = NCLASS; n < 48; ++n) Bt[n * 256 + t] = (__bf16)0.0f;
    }

    v8f acc[3] = {};

    for (int kt = 0; kt < HIDDEN; kt += 32) {
        // ---- stage A tile 128x32 f32: 1024 16B chunks, 4 per thread ----
        #pragma unroll
        for (int i = 0; i < 4; ++i) {
            int slot = t + i * 256;         // 0..1023
            int r = slot >> 3, cg = slot & 7;
            if (!GUARD) {
                unsigned lds = (unsigned)(uintptr_t)&AsubF[r * 32 + cg * 4];
                unsigned go  = (unsigned)(((unsigned)(m0 + r) * HIDDEN + kt + cg * 4)
                                          * sizeof(float));
                async_b128_to_lds(lds, go, H);
            } else {
                int gm = m0 + r;
                v4f v = {};
                if (gm < N_NODES) v = *(const v4f*)&H[(size_t)gm * HIDDEN + kt + cg * 4];
                *(v4f*)&AsubF[r * 32 + cg * 4] = v;
            }
        }
        if (!GUARD) wait_asynccnt0();
        __syncthreads();

        v16bf a = build_a_frag<true>(&AsubF[(wave * 16 + lmod) * 32], kA0);  // fused ReLU
        #pragma unroll
        for (int j = 0; j < 3; ++j) {
            int n = j * 16 + lmod;
            v16bf b = *(const v16bf*)&Bt[n * 256 + kt + kB0];  // 32B-aligned
            acc[j] = __builtin_amdgcn_wmma_f32_16x16x32_bf16(
                false, a, false, b, (short)0, acc[j], false, false);
        }
        __syncthreads();
    }

    const int rbase = m0 + wave * 16 + lhi * 8;
    #pragma unroll
    for (int j = 0; j < 3; ++j) {
        int n = j * 16 + lmod;
        if (n < NCLASS) {
            #pragma unroll
            for (int r = 0; r < 8; ++r) {
                int gm = rbase + r;
                if (!GUARD || gm < N_NODES) C[(size_t)gm * NCLASS + n] = acc[j][r];
            }
        }
    }
}

// ---------------------------------------------------------------------------
// SpMM, D=256: one block per edge, one lane per feature. Row reads are 1KB
// coalesced; h0 (102MB) and agg (102MB) both fit in the 192MB L2, so gather
// and atomic traffic stay on-chip.
// ---------------------------------------------------------------------------
__global__ __launch_bounds__(256)
void spmm_d256(const int* __restrict__ erow, const int* __restrict__ ecol,
               const float* __restrict__ evals, const float* __restrict__ h,
               float* __restrict__ out, int n_edges) {
    int e = blockIdx.x;
    if (e >= n_edges) return;
    int r = erow[e];
    int c = ecol[e];
    float v = evals[e];
    int d = threadIdx.x;
    atomicAdd(&out[(size_t)r * HIDDEN + d], v * h[(size_t)c * HIDDEN + d]);
}

// SpMM, D=40: 64 lanes per edge (40 active), 4 edges per block.
__global__ __launch_bounds__(256)
void spmm_d40(const int* __restrict__ erow, const int* __restrict__ ecol,
              const float* __restrict__ evals, const float* __restrict__ h,
              float* __restrict__ out, int n_edges) {
    int e = blockIdx.x * 4 + (threadIdx.x >> 6);
    int d = threadIdx.x & 63;
    if (e >= n_edges || d >= NCLASS) return;
    int r = erow[e];
    int c = ecol[e];
    float v = evals[e];
    atomicAdd(&out[(size_t)r * NCLASS + d], v * h[(size_t)c * NCLASS + d]);
}

// ---------------------------------------------------------------------------
// out = log_softmax(relu(a2)) per node (40 classes kept in registers)
// ---------------------------------------------------------------------------
__global__ __launch_bounds__(256)
void relu_logsoftmax40(const float* __restrict__ in, float* __restrict__ out,
                       int n_nodes) {
    int node = blockIdx.x * blockDim.x + threadIdx.x;
    if (node >= n_nodes) return;
    float vals[NCLASS];
    float m = -INFINITY;
    #pragma unroll
    for (int d = 0; d < NCLASS; ++d) {
        float v = fmaxf(in[(size_t)node * NCLASS + d], 0.0f);
        vals[d] = v;
        m = fmaxf(m, v);
    }
    float s = 0.0f;
    #pragma unroll
    for (int d = 0; d < NCLASS; ++d) s += __expf(vals[d] - m);
    float lse = m + __logf(s);
    #pragma unroll
    for (int d = 0; d < NCLASS; ++d)
        out[(size_t)node * NCLASS + d] = vals[d] - lse;
}

// ---------------------------------------------------------------------------
extern "C" void kernel_launch(void* const* d_in, const int* in_sizes, int n_in,
                              void* d_out, int out_size, void* d_ws, size_t ws_size,
                              hipStream_t stream) {
    const float* x     = (const float*)d_in[0];
    const int*   erow  = (const int*)d_in[1];
    const int*   ecol  = (const int*)d_in[2];
    const float* evals = (const float*)d_in[3];
    const float* W1    = (const float*)d_in[4];
    const float* W2    = (const float*)d_in[5];
    float*       out   = (float*)d_out;
    const int n_edges  = in_sizes[1];

    // Workspace layout (needs 2 * N_NODES * 256 * 4 bytes ~= 205 MB):
    //   bufA: h0 [N,256]  -> later reused for h2 [N,40]
    //   bufB: agg1 [N,256]-> later reused for agg2 [N,40]
    float* bufA = (float*)d_ws;
    float* bufB = bufA + (size_t)N_NODES * HIDDEN;

    // zero agg1 before atomic accumulation
    hipMemsetAsync(bufB, 0, (size_t)N_NODES * HIDDEN * sizeof(float), stream);

    // h0 = X @ W1 : full blocks (async staging) + one guarded tail block
    const int full1 = N_NODES / 64;                  // 1562
    gemm1_xw1_wmma<false><<<full1, 256, 0, stream>>>(x, W1, bufA, 0);
    gemm1_xw1_wmma<true ><<<1,     256, 0, stream>>>(x, W1, bufA, full1);

    // agg1 = segment_sum(edges, h0)
    spmm_d256<<<n_edges, 256, 0, stream>>>(erow, ecol, evals, bufA, bufB, n_edges);

    // h2 = relu(agg1) @ W2 -> bufA
    const int full2 = N_NODES / 128;                 // 781
    gemm2_hw2_wmma<false><<<full2, 256, 0, stream>>>(bufB, W2, bufA, 0);
    gemm2_hw2_wmma<true ><<<1,     256, 0, stream>>>(bufB, W2, bufA, full2);

    // zero agg2, then agg2 = segment_sum(edges, h2)
    hipMemsetAsync(bufB, 0, (size_t)N_NODES * NCLASS * sizeof(float), stream);
    spmm_d40<<<(n_edges + 3) / 4, 256, 0, stream>>>(erow, ecol, evals, bufA, bufB, n_edges);

    // out = log_softmax(relu(agg2))
    relu_logsoftmax40<<<(N_NODES + 255) / 256, 256, 0, stream>>>(bufB, out, N_NODES);
}
